// GeometricTransformerBlock_31086973288668
// MI455X (gfx1250) — matrix operand order repaired and checked by reading, not compiled, not run on hardware
//
#include <hip/hip_runtime.h>
#include <cmath>

// ---------------------------------------------------------------------------
// GeometricTransformerBlock for MI455X (gfx1250, wave32, WMMA bf16)
// B=2 N=2048 D=768 H=12 HD=64 FF=3072, radius mask on 3D coords.
// Compute-bound (~110 GFLOP over ~50MB resident, fits in 192MB L2) ->
// all matmuls through v_wmma_f32_16x16x32_bf16 with f32 accumulation.
// GEMM: 32x64 tile per wave (8 WMMAs per 32-wide k-step, ~21 FLOP/B).
// Attention: flash-style with LDS-staged V chunks.
// ---------------------------------------------------------------------------

#define Bc   2
#define Nc   2048
#define Dc   768
#define Hc   12
#define HDc  64
#define FFc  3072
#define Mc   (Bc * Nc)        // 4096 tokens

typedef __attribute__((ext_vector_type(16))) __bf16 v16bf;
typedef __attribute__((ext_vector_type(8)))  __bf16 v8bf;
typedef __attribute__((ext_vector_type(8)))  float  v8f;

static __device__ __forceinline__ __bf16 f2bf(float f) {
    unsigned u = __float_as_uint(f);
    u += 0x7FFFu + ((u >> 16) & 1u);           // round-to-nearest-even
    unsigned short s = (unsigned short)(u >> 16);
    return __builtin_bit_cast(__bf16, s);
}

static __device__ __forceinline__ v16bf cat8(v8bf lo, v8bf hi) {
    return __builtin_shufflevector(lo, hi, 0,1,2,3,4,5,6,7,8,9,10,11,12,13,14,15);
}

static __device__ __forceinline__ v8f wmma_bf(v16bf a, v16bf b, v8f c) {
    // D = A(16x32 bf16) x B(32x16 bf16) + C(16x16 f32)
    return __builtin_amdgcn_wmma_f32_16x16x32_bf16(false, a, false, b, (short)0, c,
                                                   false, false);
}

// ---------------------------------------------------------------------------
// Weight convert + transpose: w[K][N] f32 -> wT[N][K] bf16 (K-major columns so
// WMMA B-fragments are 16 contiguous bf16 = one 32B load per lane).
// ---------------------------------------------------------------------------
__global__ __launch_bounds__(256)
void cvt_t_kernel(const float* __restrict__ w, __bf16* __restrict__ wT, int K, int Nn) {
    size_t i = (size_t)blockIdx.x * 256 + threadIdx.x;
    if (i >= (size_t)K * Nn) return;
    int n = (int)(i % Nn);
    int k = (int)(i / Nn);
    wT[(size_t)n * K + k] = f2bf(w[i]);
}

// ---------------------------------------------------------------------------
// LayerNorm over D=768, one block (256 thr) per token, bf16 output.
// ---------------------------------------------------------------------------
__global__ __launch_bounds__(256)
void ln_bf16_kernel(const float* __restrict__ x, const float* __restrict__ gam,
                    const float* __restrict__ bet, __bf16* __restrict__ o) {
    __shared__ float red[256];
    const int t = threadIdx.x;
    const float* xp = x + (size_t)blockIdx.x * Dc;
    float s = 0.f;
    for (int i = t; i < Dc; i += 256) s += xp[i];
    red[t] = s; __syncthreads();
    for (int off = 128; off; off >>= 1) {
        if (t < off) red[t] += red[t + off];
        __syncthreads();
    }
    float mean = red[0] * (1.0f / Dc);
    __syncthreads();
    float v = 0.f;
    for (int i = t; i < Dc; i += 256) { float d = xp[i] - mean; v += d * d; }
    red[t] = v; __syncthreads();
    for (int off = 128; off; off >>= 1) {
        if (t < off) red[t] += red[t + off];
        __syncthreads();
    }
    float rstd = rsqrtf(red[0] * (1.0f / Dc) + 1e-5f);
    __bf16* op = o + (size_t)blockIdx.x * Dc;
    for (int i = t; i < Dc; i += 256)
        op[i] = f2bf((xp[i] - mean) * rstd * gam[i] + bet[i]);
}

// ---------------------------------------------------------------------------
// GEMM: C[M,Nn] = A[M,K](bf16) @ WT[Nn,K]^T(bf16) + bias.
// One wave per 32x64 tile: 2 A-fragments (row blocks) share each of 4
// B-fragments -> 8 WMMAs per 32-wide k-step (~21 FLOP/B from L2).
// 8 waves / 256-thread block.
// mode 0: bf16 out (QKV)    mode 1: f32 out + residual (proj / FF2)
// mode 2: exact-GELU -> bf16 out (FF1)
// ---------------------------------------------------------------------------
__global__ __launch_bounds__(256)
void gemm_kernel(const __bf16* __restrict__ A, const __bf16* __restrict__ WT,
                 const float* __restrict__ bias, const float* __restrict__ res,
                 float* __restrict__ oF, __bf16* __restrict__ oB,
                 int M, int Nn, int K, int mode) {
    const int lane = threadIdx.x & 31;
    const int wave = threadIdx.x >> 5;
    const int g  = lane >> 4;        // half-wave group
    const int hl = lane & 15;
    const int ntiles = Nn >> 6;
    int wid = blockIdx.x * 8 + wave;
    int mt = wid / ntiles;
    int nt = wid - mt * ntiles;
    if (mt * 32 >= M) return;
    const int m0 = mt * 32, n0 = nt * 64;

    v8f acc[8] = {};   // [s] rows m0..m0+15, [4+s] rows m0+16..m0+31
    // A fragment: row = m0(+16)+hl, K elements e+16g -> two 16B runs
    const __bf16* Ar0 = A + (size_t)(m0 + hl) * K + (g ? 16 : 0);
    const __bf16* Ar1 = Ar0 + (size_t)16 * K;
    // B fragment: col = n0+s*16+hl, K elements e+16g -> one 32B run
    const __bf16* W0 = WT + (size_t)(n0 + hl) * K + (g ? 16 : 0);
    for (int k0 = 0; k0 < K; k0 += 32) {
        v16bf af0 = cat8(*(const v8bf*)(Ar0 + k0), *(const v8bf*)(Ar0 + k0 + 8));
        v16bf af1 = cat8(*(const v8bf*)(Ar1 + k0), *(const v8bf*)(Ar1 + k0 + 8));
#pragma unroll
        for (int s = 0; s < 4; ++s) {
            v16bf bf = *(const v16bf*)(W0 + (size_t)(s * 16) * K + k0);
            acc[s]     = wmma_bf(af0, bf, acc[s]);
            acc[4 + s] = wmma_bf(af1, bf, acc[4 + s]);
        }
    }
#pragma unroll
    for (int s = 0; s < 4; ++s) {
        const int col = n0 + s * 16 + hl;
        const float bv = bias[col];
#pragma unroll
        for (int half = 0; half < 2; ++half) {
#pragma unroll
            for (int r = 0; r < 8; ++r) {
                const int row = m0 + half * 16 + r + 8 * g;   // C-layout row
                const size_t idx = (size_t)row * Nn + col;
                float v = acc[half * 4 + s][r] + bv;
                if (mode == 0) {
                    oB[idx] = f2bf(v);
                } else if (mode == 1) {
                    oF[idx] = v + res[idx];
                } else {
                    float t = 0.5f * v * (1.0f + erff(v * 0.70710678118654752f));
                    oB[idx] = f2bf(t);
                }
            }
        }
    }
}

// ---------------------------------------------------------------------------
// Flash-style masked attention. qkv bf16 layout [B*N][3][H][64] (row = 2304).
// One wave per (b, h, 16-row q-tile); loop over 32-key chunks.
//   scores: 4 WMMAs (2 col tiles x K=64)       P@V: 4 WMMAs (HD=64)
// Online softmax: row data lives across 16 lanes in C-layout -> __shfl_xor
// reductions within half-wave; P staged through LDS to re-layout C -> A;
// V chunk staged into LDS with vectorized loads, fragments gathered from LDS.
// ---------------------------------------------------------------------------
#define AW 4
__global__ __launch_bounds__(AW * 32)
void attn_kernel(const __bf16* __restrict__ qkv, const float* __restrict__ coords,
                 __bf16* __restrict__ o) {
    const int lane = threadIdx.x & 31;
    const int wave = threadIdx.x >> 5;
    const int g  = lane >> 4;
    const int hl = lane & 15;
    const int kb = g ? 16 : 0;

    __shared__ alignas(32) __bf16 Pld[AW][16][32];
    __shared__ alignas(32) __bf16 Vld[AW][32][64];
    __shared__ float qc[AW][16][3];
    __shared__ float kcs[AW][32][3];

    const int qtiles = Nc / 16;
    int wid = blockIdx.x * AW + wave;
    int b   = wid / (Hc * qtiles);
    int rem = wid - b * (Hc * qtiles);
    int h   = rem / qtiles;
    int q0  = (rem - h * qtiles) * 16;

    const size_t RS = 3 * (size_t)Dc;                    // 2304 elems/token
    const __bf16* base = qkv + (size_t)b * Nc * RS;

    if (lane < 16) {
        const float* cp = coords + ((size_t)b * Nc + q0 + lane) * 3;
        qc[wave][lane][0] = cp[0];
        qc[wave][lane][1] = cp[1];
        qc[wave][lane][2] = cp[2];
    }

    // Q A-fragments (hd 0..31 and 32..63), loaded once per q-tile.
    v16bf qf0, qf1;
    {
        const __bf16* qp = base + (size_t)(q0 + hl) * RS + (size_t)h * HDc;
        qf0 = cat8(*(const v8bf*)(qp + kb),      *(const v8bf*)(qp + kb + 8));
        qf1 = cat8(*(const v8bf*)(qp + 32 + kb), *(const v8bf*)(qp + 32 + kb + 8));
    }

    float mrun[8], lrun[8];
    v8f accO[4] = {};
#pragma unroll
    for (int r = 0; r < 8; ++r) { mrun[r] = -INFINITY; lrun[r] = 0.f; }

    for (int k0 = 0; k0 < Nc; k0 += 32) {
        {   // stage 32 key coords (1 key per lane)
            const float* cp = coords + ((size_t)b * Nc + k0 + lane) * 3;
            kcs[wave][lane][0] = cp[0];
            kcs[wave][lane][1] = cp[1];
            kcs[wave][lane][2] = cp[2];
        }
        {   // stage V chunk [32 keys][64 hd] with vectorized 16B loads
            const __bf16* vp = base + (size_t)(k0 + lane) * RS + 2 * Dc
                             + (size_t)h * HDc;
#pragma unroll
            for (int j = 0; j < 8; ++j)
                *(v8bf*)&Vld[wave][lane][j * 8] = *(const v8bf*)(vp + j * 8);
        }
        __syncthreads();

        // scores for two 16-col tiles; K^T B-fragments are contiguous 32B loads
        v8f s0 = {}, s1 = {};
        {
            const __bf16* kp0 = base + (size_t)(k0 + hl) * RS + Dc
                              + (size_t)h * HDc + (g ? 16 : 0);
            const __bf16* kp1 = kp0 + 16 * RS;
            s0 = wmma_bf(qf0, *(const v16bf*)kp0,        s0);
            s0 = wmma_bf(qf1, *(const v16bf*)(kp0 + 32), s0);
            s1 = wmma_bf(qf0, *(const v16bf*)kp1,        s1);
            s1 = wmma_bf(qf1, *(const v16bf*)(kp1 + 32), s1);
        }

        const float kx0 = kcs[wave][hl][0],      ky0 = kcs[wave][hl][1],      kz0 = kcs[wave][hl][2];
        const float kx1 = kcs[wave][16 + hl][0], ky1 = kcs[wave][16 + hl][1], kz1 = kcs[wave][16 + hl][2];

        float sc[8];
#pragma unroll
        for (int r = 0; r < 8; ++r) {
            const int qr = r + 8 * g;
            const float qx = qc[wave][qr][0], qy = qc[wave][qr][1], qz = qc[wave][qr][2];
            float dx = qx - kx0, dy = qy - ky0, dz = qz - kz0;
            float d2 = dx * dx + dy * dy + dz * dz;
            float a0 = (d2 <= 4.0f) ? s0[r] * 0.125f : -1.0e9f;   // 1/sqrt(64)
            dx = qx - kx1; dy = qy - ky1; dz = qz - kz1;
            d2 = dx * dx + dy * dy + dz * dz;
            float a1 = (d2 <= 4.0f) ? s1[r] * 0.125f : -1.0e9f;

            float t = fmaxf(a0, a1);
#pragma unroll
            for (int off = 1; off < 16; off <<= 1)
                t = fmaxf(t, __shfl_xor(t, off, 32));
            float mnew = fmaxf(mrun[r], t);
            float scale_old = __expf(mrun[r] - mnew);              // exp(-inf)=0
            mrun[r] = mnew;
            float p0 = __expf(a0 - mnew);
            float p1 = __expf(a1 - mnew);
            float rs = p0 + p1;
#pragma unroll
            for (int off = 1; off < 16; off <<= 1)
                rs += __shfl_xor(rs, off, 32);
            lrun[r] = lrun[r] * scale_old + rs;
            sc[r] = scale_old;
            Pld[wave][qr][hl]      = f2bf(p0);
            Pld[wave][qr][16 + hl] = f2bf(p1);
        }
#pragma unroll
        for (int s = 0; s < 4; ++s)
#pragma unroll
            for (int r = 0; r < 8; ++r) accO[s][r] *= sc[r];
        __syncthreads();

        // P back from LDS in A-fragment layout
        v16bf pf;
        {
            const __bf16* pp = &Pld[wave][hl][0];
            pf = cat8(*(const v8bf*)(pp + kb), *(const v8bf*)(pp + kb + 8));
        }
        // P(16x32) @ V(32x64): 4 WMMAs, V B-fragments gathered from LDS
#pragma unroll
        for (int s = 0; s < 4; ++s) {
            v16bf vf;
#pragma unroll
            for (int e = 0; e < 16; ++e)
                vf[e] = Vld[wave][e + 16 * g][s * 16 + hl];
            accO[s] = wmma_bf(pf, vf, accO[s]);
        }
    }

    // epilogue: out[q, h*64+hd] = acc/l  (bf16 for out-projection GEMM)
#pragma unroll
    for (int s = 0; s < 4; ++s)
#pragma unroll
        for (int r = 0; r < 8; ++r) {
            int q = q0 + r + 8 * g;
            float oval = accO[s][r] / lrun[r];
            o[((size_t)b * Nc + q) * Dc + (size_t)h * HDc + s * 16 + hl] = f2bf(oval);
        }
}

// ---------------------------------------------------------------------------
// Launch
// ---------------------------------------------------------------------------
extern "C" void kernel_launch(void* const* d_in, const int* in_sizes, int n_in,
                              void* d_out, int out_size, void* d_ws, size_t ws_size,
                              hipStream_t stream) {
    (void)in_sizes; (void)n_in; (void)out_size; (void)ws_size;

    const float* x      = (const float*)d_in[0];
    const float* coords = (const float*)d_in[1];
    const float* w_qkv  = (const float*)d_in[2];
    const float* b_qkv  = (const float*)d_in[3];
    const float* w_out  = (const float*)d_in[4];
    const float* b_out  = (const float*)d_in[5];
    const float* w_ff1  = (const float*)d_in[6];
    const float* b_ff1  = (const float*)d_in[7];
    const float* w_ff2  = (const float*)d_in[8];
    const float* b_ff2  = (const float*)d_in[9];
    const float* g1     = (const float*)d_in[10];
    const float* be1    = (const float*)d_in[11];
    const float* g2     = (const float*)d_in[12];
    const float* be2    = (const float*)d_in[13];
    float* out = (float*)d_out;

    // workspace partition (256B aligned). h_bf aliases the dead qkv region.
    char* p = (char*)d_ws;
    auto take = [&](size_t bytes) {
        char* r = p;
        p += (bytes + 255) & ~(size_t)255;
        return r;
    };
    __bf16* wqkvT = (__bf16*)take((size_t)Dc * 3 * Dc * 2);   //  3.5 MB
    __bf16* woutT = (__bf16*)take((size_t)Dc * Dc * 2);       //  1.2 MB
    __bf16* wff1T = (__bf16*)take((size_t)Dc * FFc * 2);      //  4.7 MB
    __bf16* wff2T = (__bf16*)take((size_t)FFc * Dc * 2);      //  4.7 MB
    __bf16* xn    = (__bf16*)take((size_t)Mc * Dc * 2);       //  6.3 MB (ln1 then ln2)
    __bf16* regA  = (__bf16*)take((size_t)Mc * FFc * 2);      // 25.2 MB (qkv, then h)
    __bf16* attnb = (__bf16*)take((size_t)Mc * Dc * 2);       //  6.3 MB
    float*  x2    = (float*) take((size_t)Mc * Dc * 4);       // 12.6 MB
    __bf16* qkvb  = regA;
    __bf16* hb    = regA;

    auto cdiv = [](size_t a, size_t b) { return (int)((a + b - 1) / b); };

    // 1) weight convert+transpose
    cvt_t_kernel<<<cdiv((size_t)Dc * 3 * Dc, 256), 256, 0, stream>>>(w_qkv, wqkvT, Dc, 3 * Dc);
    cvt_t_kernel<<<cdiv((size_t)Dc * Dc, 256),     256, 0, stream>>>(w_out, woutT, Dc, Dc);
    cvt_t_kernel<<<cdiv((size_t)Dc * FFc, 256),    256, 0, stream>>>(w_ff1, wff1T, Dc, FFc);
    cvt_t_kernel<<<cdiv((size_t)FFc * Dc, 256),    256, 0, stream>>>(w_ff2, wff2T, FFc, Dc);

    // 2) LN1 -> xn (bf16)
    ln_bf16_kernel<<<Mc, 256, 0, stream>>>(x, g1, be1, xn);

    // 3) QKV: xn @ w_qkv + b_qkv -> qkv bf16 [M, 2304]
    gemm_kernel<<<(Mc / 32) * (3 * Dc / 64) / 8, 256, 0, stream>>>(
        xn, wqkvT, b_qkv, nullptr, nullptr, qkvb, Mc, 3 * Dc, Dc, 0);

    // 4) masked flash attention -> attnb bf16 [M, 768]
    attn_kernel<<<(Bc * Hc * (Nc / 16)) / AW, AW * 32, 0, stream>>>(qkvb, coords, attnb);

    // 5) out-proj + residual: x2 = attnb @ w_out + b_out + x   (f32)
    gemm_kernel<<<(Mc / 32) * (Dc / 64) / 8, 256, 0, stream>>>(
        attnb, woutT, b_out, x, x2, nullptr, Mc, Dc, Dc, 1);

    // 6) LN2 -> xn (bf16, reused)
    ln_bf16_kernel<<<Mc, 256, 0, stream>>>(x2, g2, be2, xn);

    // 7) FF1 + GELU -> hb bf16 [M, 3072]
    gemm_kernel<<<(Mc / 32) * (FFc / 64) / 8, 256, 0, stream>>>(
        xn, wff1T, b_ff1, nullptr, nullptr, hb, Mc, FFc, Dc, 2);

    // 8) FF2 + residual -> d_out f32
    gemm_kernel<<<(Mc / 32) * (Dc / 64) / 8, 256, 0, stream>>>(
        hb, wff2T, b_ff2, x2, out, nullptr, Mc, Dc, FFc, 1);
}
